// Model_73100343377962
// MI455X (gfx1250) — matrix operand order repaired and checked by reading, not compiled, not run on hardware
//
#include <hip/hip_runtime.h>
#include <hip/hip_bf16.h>
#include <stdint.h>

#define SQ 512
#define NB 64
#define INP 64
#define HH 256

typedef __attribute__((ext_vector_type(16))) __bf16 v16bf;
typedef __attribute__((ext_vector_type(8)))  float  v8f;
typedef __attribute__((ext_vector_type(8)))  unsigned short u16x8;
typedef __attribute__((ext_vector_type(16))) unsigned short u16x16;

static __device__ __forceinline__ unsigned short f2bf(float f) {
    unsigned u = __float_as_uint(f);
    u += 0x7fffu + ((u >> 16) & 1u);
    return (unsigned short)(u >> 16);
}
static __device__ __forceinline__ float bf2f(unsigned short h) {
    return __uint_as_float((unsigned)h << 16);
}
static __device__ __forceinline__ float2 bf2x2(unsigned e) {
    float2 r;
    r.x = __uint_as_float(e << 16);
    r.y = __uint_as_float(e & 0xffff0000u);
    return r;
}
static __device__ __forceinline__ float sigm(float x) {
    return 1.0f / (1.0f + __expf(-x));
}

struct A32 { u16x8 lo, hi; };

static __device__ __forceinline__ v16bf load_afrag(const unsigned short* p) {
    A32 t;
    t.lo = *(const u16x8*)p;
    t.hi = *(const u16x8*)(p + 16);
    return __builtin_bit_cast(v16bf, t);
}
static __device__ __forceinline__ v16bf load_bfrag(const unsigned short* p) {
    u16x16 t = *(const u16x16*)p;
    return __builtin_bit_cast(v16bf, t);
}

// D = A(64 x Ktot row-major bf16) tile (mt) x packed-B tiles (nt), K loop
static __device__ __forceinline__ v8f wmma_rowcol(const unsigned short* A, int ldk, int kt_cnt,
                                                  const unsigned short* Bt, int ntiles,
                                                  int nt, int mt, int lane) {
    v8f acc = {};
    const int m = mt * 16 + (lane & 15);
    const int halfk = (lane >> 4) * 8;
    const unsigned short* arow = A + (size_t)m * ldk + halfk;
    for (int kt = 0; kt < kt_cnt; ++kt) {
        v16bf a = load_afrag(arow + kt * 32);
        v16bf b = load_bfrag(Bt + ((size_t)(kt * ntiles + nt) * 512) + lane * 16);
        acc = __builtin_amdgcn_wmma_f32_16x16x32_bf16(false, a, false, b, (short)0, acc, false, false);
    }
    return acc;
}

static __device__ __forceinline__ void store_acc(float* G, int ldg, int mt, int nt, int lane, v8f acc) {
    const int gm = mt * 16 + (lane >> 4) * 8;
    const int gn = nt * 16 + (lane & 15);
#pragma unroll
    for (int r = 0; r < 8; ++r) G[(size_t)(gm + r) * ldg + gn] = acc[r];
}

static __device__ __forceinline__ void grid_barrier(volatile unsigned* bar, unsigned nblk) {
    __syncthreads();
    if (threadIdx.x == 0) {
        __threadfence();
        unsigned gen = bar[1];
        if (atomicAdd((unsigned*)&bar[0], 1u) == nblk - 1u) {
            bar[0] = 0u;
            __threadfence();
            atomicAdd((unsigned*)&bar[1], 1u);
        } else {
            while (bar[1] == gen) { __builtin_amdgcn_s_sleep(2); }
        }
        __threadfence();
    }
    __syncthreads();
}

// ---------------- weight packing + bias fusion ----------------
__global__ void k_pack(const float* __restrict__ Wih0, const float* __restrict__ Whh0,
                       const float* __restrict__ bih0, const float* __restrict__ bhh0,
                       const float* __restrict__ Wih1, const float* __restrict__ Whh1,
                       const float* __restrict__ bih1, const float* __restrict__ bhh1,
                       const float* __restrict__ Wihd, const float* __restrict__ Whhd,
                       const float* __restrict__ bihd, const float* __restrict__ bhhd,
                       unsigned short* __restrict__ Wpk0, unsigned short* __restrict__ Wpk1,
                       unsigned short* __restrict__ WpkD,
                       float* __restrict__ bias0, float* __restrict__ bias1, float* __restrict__ biasD) {
    const long R0 = 2L * 320 * 1024;
    const long R1 = 2L * 768 * 1024;
    const long RD = 1600L * 2048;
    const long total = R0 + R1 + RD + 2048 + 2048 + 2048;
    for (long i = (long)blockIdx.x * blockDim.x + threadIdx.x; i < total;
         i += (long)gridDim.x * blockDim.x) {
        long r = i;
        if (r < R0) {
            int d = (int)(r / (320 * 1024)); int rem = (int)(r % (320 * 1024));
            int n = rem / 320, k = rem % 320;
            float v = (k < 64) ? Wih0[((size_t)d * 1024 + n) * 64 + k]
                               : Whh0[((size_t)d * 1024 + n) * 256 + (k - 64)];
            Wpk0[(size_t)d * (10 * 64 * 512) + ((size_t)((k >> 5) * 64 + (n >> 4)) * 512)
                 + (((k >> 3) & 1) * 16 + (n & 15)) * 16 + (k & 7) + ((k >> 4) & 1) * 8] = f2bf(v);
            continue;
        }
        r -= R0;
        if (r < R1) {
            int d = (int)(r / (768 * 1024)); int rem = (int)(r % (768 * 1024));
            int n = rem / 768, k = rem % 768;
            float v = (k < 512) ? Wih1[((size_t)d * 1024 + n) * 512 + k]
                                : Whh1[((size_t)d * 1024 + n) * 256 + (k - 512)];
            Wpk1[(size_t)d * (24 * 64 * 512) + ((size_t)((k >> 5) * 64 + (n >> 4)) * 512)
                 + (((k >> 3) & 1) * 16 + (n & 15)) * 16 + (k & 7) + ((k >> 4) & 1) * 8] = f2bf(v);
            continue;
        }
        r -= R1;
        if (r < RD) {
            int n = (int)(r / 1600), k = (int)(r % 1600);
            float v = (k < 1088) ? Wihd[(size_t)n * 1088 + k]
                                 : Whhd[(size_t)n * 512 + (k - 1088)];
            WpkD[((size_t)((k >> 5) * 128 + (n >> 4)) * 512) + (((k >> 3) & 1) * 16 + (n & 15)) * 16 + (k & 7) + ((k >> 4) & 1) * 8] = f2bf(v);
            continue;
        }
        r -= RD;
        if (r < 2048)      { bias0[r] = bih0[r] + bhh0[r]; continue; }
        r -= 2048;
        if (r < 2048)      { bias1[r] = bih1[r] + bhh1[r]; continue; }
        r -= 2048;
        biasD[r] = bihd[r] + bhhd[r];
    }
}

// ---------------- bidirectional LSTM layer (persistent) ----------------
__global__ void __launch_bounds__(256) k_lstm(
    const float* __restrict__ x, const unsigned short* __restrict__ hin,
    int Kin, int Ktot,
    const unsigned short* __restrict__ Wpk, long pkstride,
    const float* __restrict__ bias,
    unsigned short* __restrict__ hout,
    float* __restrict__ hT, float* __restrict__ cT,
    unsigned short* __restrict__ Abuf, float* __restrict__ Gbuf,
    float* __restrict__ hst, float* __restrict__ cst,
    unsigned* bar, int nblk, int nbd) {
    const int dir = blockIdx.x / nbd;
    const int bid = blockIdx.x % nbd;
    const int tidd = bid * (int)blockDim.x + (int)threadIdx.x;
    const int nthr = nbd * (int)blockDim.x;
    const int lane = threadIdx.x & 31;
    const int wvd = tidd >> 5;
    const int nwv = nthr >> 5;

    unsigned short* A = Abuf + (size_t)dir * 64 * Ktot;
    float* G  = Gbuf + (size_t)dir * 64 * 1024;
    float* hS = hst + dir * 64 * HH;
    float* cS = cst + dir * 64 * HH;
    const unsigned short* W = Wpk + (size_t)dir * pkstride;
    const float* bs = bias + dir * 1024;

    for (int i = tidd; i < 64 * Ktot; i += nthr) A[i] = 0;
    for (int i = tidd; i < 64 * HH; i += nthr) { hS[i] = 0.f; cS[i] = 0.f; }
    grid_barrier(bar, nblk);

    const int kt_cnt = Ktot >> 5;
    const int tpw = 256 / nwv;  // tiles per wave

    for (int t = 0; t < SQ; ++t) {
        const int tsrc = (dir == 0) ? t : (SQ - 1 - t);
        if (x) {
            for (int i = tidd; i < 64 * Kin; i += nthr) {
                int b = i / Kin, k = i - b * Kin;
                A[(size_t)b * Ktot + k] = f2bf(x[((size_t)b * SQ + tsrc) * INP + k]);
            }
        } else {
            for (int i = tidd; i < 64 * Kin; i += nthr) {
                int b = i / Kin, k = i - b * Kin;
                A[(size_t)b * Ktot + k] = hin[((size_t)tsrc * 64 + b) * 512 + k];
            }
        }
        grid_barrier(bar, nblk);
        for (int qq = 0; qq < tpw; ++qq) {
            int tt = wvd * tpw + qq;
            int mt = tt >> 6, nt = tt & 63;
            v8f acc = wmma_rowcol(A, Ktot, kt_cnt, W, 64, nt, mt, lane);
            store_acc(G, 1024, mt, nt, lane, acc);
        }
        grid_barrier(bar, nblk);
        for (int i = tidd; i < 64 * HH; i += nthr) {
            int b = i >> 8, n = i & 255;
            const float* g = G + (size_t)b * 1024;
            float ig = g[n] + bs[n];
            float fg = g[256 + n] + bs[256 + n];
            float gg = g[512 + n] + bs[512 + n];
            float og = g[768 + n] + bs[768 + n];
            float c0 = cS[i];
            float c1 = sigm(fg) * c0 + sigm(ig) * tanhf(gg);
            float h1 = sigm(og) * tanhf(c1);
            cS[i] = c1; hS[i] = h1;
            unsigned short hb = f2bf(h1);
            A[(size_t)b * Ktot + Kin + n] = hb;
            hout[((size_t)tsrc * 64 + b) * 512 + (size_t)dir * 256 + n] = hb;
        }
        grid_barrier(bar, nblk);
    }
    if (hT) {
        for (int i = tidd; i < 64 * HH; i += nthr) {
            int b = i >> 8, n = i & 255;
            hT[(size_t)b * 512 + dir * 256 + n] = hS[i];
            cT[(size_t)b * 512 + dir * 256 + n] = cS[i];
        }
    }
}

// ---------------- attention decoder (persistent, 64 blocks x 512 thr) ----------------
__global__ void __launch_bounds__(512) k_dec(
    const float* __restrict__ x, const unsigned short* __restrict__ enc,
    const unsigned short* __restrict__ Wpk, const float* __restrict__ bias,
    const float* __restrict__ h0c0,
    unsigned short* __restrict__ A, float* __restrict__ G,
    float* __restrict__ hst, float* __restrict__ cst,
    unsigned short* __restrict__ dec_hs,
    unsigned* bar, int nblk) {
    __shared__ float sc[2][512];
    const int tid = blockIdx.x * 512 + threadIdx.x;   // 0..32767
    const int nthr = nblk * 512;
    const int lane = threadIdx.x & 31;
    const int wv = threadIdx.x >> 5;                  // 0..15
    const int rl = wv >> 3;                           // 0..1
    const int wsub = wv & 7;                          // 0..7
    const int row = blockIdx.x * 2 + rl;              // 0..127
    const int q = row >> 6;
    const int b = row & 63;
    const int wvg = blockIdx.x * 16 + wv;             // 0..1023

    for (int i = tid; i < 64 * 512; i += nthr) {
        float hv = h0c0[i], cv = h0c0[64 * 512 + i];
        hst[i] = hv; cst[i] = cv;
        int bb = i >> 9, n = i & 511;
        A[(size_t)bb * 1600 + 1088 + n] = f2bf(hv);
    }
    grid_barrier(bar, nblk);

    for (int t = 0; t < SQ; ++t) {
        // x slice into A
        for (int i = tid; i < 64 * 64; i += nthr) {
            int bb = i >> 6, k = i & 63;
            A[(size_t)bb * 1600 + k] = f2bf(x[((size_t)bb * SQ + t) * INP + k]);
        }
        // scores: row (q,b), 8 waves split time; lanes split d (coalesced)
        const float* Q = (q == 0 ? hst : cst) + (size_t)b * 512;
        float2 qr[8];
#pragma unroll
        for (int j = 0; j < 8; ++j) qr[j] = *(const float2*)(Q + 2 * (lane + 32 * j));
        for (int tp = wsub * 64; tp < wsub * 64 + 64; ++tp) {
            const unsigned* ep = (const unsigned*)(enc + ((size_t)tp * 64 + b) * 512);
            float acc = 0.f;
#pragma unroll
            for (int j = 0; j < 8; ++j) {
                float2 ef = bf2x2(ep[lane + 32 * j]);
                acc += qr[j].x * ef.x + qr[j].y * ef.y;
            }
#pragma unroll
            for (int off = 16; off > 0; off >>= 1) acc += __shfl_xor(acc, off, 32);
            if (lane == 0) sc[rl][tp] = acc;
        }
        __syncthreads();
        // softmax (each wave redundantly reduces full row)
        float mx = -3.0e38f;
#pragma unroll
        for (int j = 0; j < 16; ++j) mx = fmaxf(mx, sc[rl][lane + 32 * j]);
#pragma unroll
        for (int off = 16; off > 0; off >>= 1) mx = fmaxf(mx, __shfl_xor(mx, off, 32));
        float sm = 0.f;
#pragma unroll
        for (int j = 0; j < 16; ++j) sm += __expf(sc[rl][lane + 32 * j] - mx);
#pragma unroll
        for (int off = 16; off > 0; off >>= 1) sm += __shfl_xor(sm, off, 32);
        float inv = 1.f / sm;
        __syncthreads();
#pragma unroll
        for (int j = 0; j < 2; ++j) {
            int tp = wsub * 64 + lane + 32 * j;
            sc[rl][tp] = __expf(sc[rl][tp] - mx) * inv;
        }
        __syncthreads();
        // context: lanes own d-pairs, loop time
        {
            int pu = wsub * 32 + lane;  // pair 0..255
            float cx = 0.f, cy = 0.f;
            const unsigned* eb = (const unsigned*)enc;
            for (int t2 = 0; t2 < 512; ++t2) {
                float p = sc[rl][t2];
                float2 ef = bf2x2(eb[((size_t)t2 * 64 + b) * 256 + pu]);
                cx += p * ef.x; cy += p * ef.y;
            }
            size_t col = (size_t)b * 1600 + 64 + (size_t)q * 512 + 2 * pu;
            A[col] = f2bf(cx);
            A[col + 1] = f2bf(cy);
        }
        grid_barrier(bar, nblk);
        // GEMM: g = A(64x1600) @ Wcat^T -> (64x2048), 512 tiles
        if (wvg < 512) {
            int mt = wvg >> 7, nt = wvg & 127;
            v8f acc = wmma_rowcol(A, 1600, 50, Wpk, 128, nt, mt, lane);
            store_acc(G, 2048, mt, nt, lane, acc);
        }
        grid_barrier(bar, nblk);
        // gates: one element per thread (64*512 == nthr)
        {
            int bb = tid >> 9, n = tid & 511;
            const float* g = G + (size_t)bb * 2048;
            float ig = g[n] + bias[n];
            float fg = g[512 + n] + bias[512 + n];
            float gg = g[1024 + n] + bias[1024 + n];
            float og = g[1536 + n] + bias[1536 + n];
            float c0 = cst[tid];
            float c1 = sigm(fg) * c0 + sigm(ig) * tanhf(gg);
            float h1 = sigm(og) * tanhf(c1);
            cst[tid] = c1; hst[tid] = h1;
            unsigned short hb = f2bf(h1);
            A[(size_t)bb * 1600 + 1088 + n] = hb;
            dec_hs[((size_t)t * 64 + bb) * 512 + n] = hb;
        }
        grid_barrier(bar, nblk);
    }
}

// ---------------- final projection ----------------
__global__ void k_proj(const unsigned short* __restrict__ dec_hs,
                       const float* __restrict__ Wp, const float* __restrict__ bp,
                       float* __restrict__ out) {
    int wv = (blockIdx.x * blockDim.x + threadIdx.x) >> 5;
    int lane = threadIdx.x & 31;
    int nwv = (gridDim.x * blockDim.x) >> 5;
    for (int o = wv; o < 64 * 512; o += nwv) {
        int s = o >> 6, b = o & 63;
        const unsigned short* hp = dec_hs + ((size_t)s * 64 + b) * 512;
        float acc = 0.f;
#pragma unroll
        for (int j = 0; j < 16; ++j) acc += bf2f(hp[lane + 32 * j]) * Wp[lane + 32 * j];
#pragma unroll
        for (int off = 16; off > 0; off >>= 1) acc += __shfl_xor(acc, off, 32);
        if (lane == 0) out[(size_t)b * 512 + s] = acc + bp[0];
    }
}

extern "C" void kernel_launch(void* const* d_in, const int* in_sizes, int n_in,
                              void* d_out, int out_size, void* d_ws, size_t ws_size,
                              hipStream_t stream) {
    (void)in_sizes; (void)n_in; (void)out_size; (void)ws_size;
    const float* x    = (const float*)d_in[0];
    const float* Wih0 = (const float*)d_in[1];
    const float* Whh0 = (const float*)d_in[2];
    const float* bih0 = (const float*)d_in[3];
    const float* bhh0 = (const float*)d_in[4];
    const float* Wih1 = (const float*)d_in[5];
    const float* Whh1 = (const float*)d_in[6];
    const float* bih1 = (const float*)d_in[7];
    const float* bhh1 = (const float*)d_in[8];
    const float* Wihd = (const float*)d_in[9];
    const float* Whhd = (const float*)d_in[10];
    const float* bihd = (const float*)d_in[11];
    const float* bhhd = (const float*)d_in[12];
    const float* Wp   = (const float*)d_in[13];
    const float* bp   = (const float*)d_in[14];

    uint8_t* p = (uint8_t*)d_ws;
    auto alloc = [&](size_t bytes) -> void* {
        void* r = (void*)p;
        p += (bytes + 255) & ~(size_t)255;
        return r;
    };
    unsigned*       bar   = (unsigned*)alloc(256);
    unsigned short* Wpk0  = (unsigned short*)alloc((size_t)2 * 10 * 64 * 512 * 2);
    unsigned short* Wpk1  = (unsigned short*)alloc((size_t)2 * 24 * 64 * 512 * 2);
    unsigned short* WpkD  = (unsigned short*)alloc((size_t)50 * 128 * 512 * 2);
    float*          bias0 = (float*)alloc(2 * 1024 * 4);
    float*          bias1 = (float*)alloc(2 * 1024 * 4);
    float*          biasD = (float*)alloc(2048 * 4);
    unsigned short* h1s   = (unsigned short*)alloc((size_t)SQ * 64 * 512 * 2);  // also dec_hs
    unsigned short* encb  = (unsigned short*)alloc((size_t)SQ * 64 * 512 * 2);
    float*          h0c0  = (float*)alloc((size_t)2 * 64 * 512 * 4);
    unsigned short* A0    = (unsigned short*)alloc((size_t)2 * 64 * 320 * 2);
    unsigned short* A1    = (unsigned short*)alloc((size_t)2 * 64 * 768 * 2);
    unsigned short* AD    = (unsigned short*)alloc((size_t)64 * 1600 * 2);
    float*          Gbuf  = (float*)alloc((size_t)2 * 64 * 1024 * 4);  // == 64*2048 for decoder
    float*          hstL  = (float*)alloc((size_t)2 * 64 * 256 * 4);
    float*          cstL  = (float*)alloc((size_t)2 * 64 * 256 * 4);
    float*          hstD  = (float*)alloc((size_t)64 * 512 * 4);
    float*          cstD  = (float*)alloc((size_t)64 * 512 * 4);

    k_pack<<<2048, 256, 0, stream>>>(Wih0, Whh0, bih0, bhh0, Wih1, Whh1, bih1, bhh1,
                                     Wihd, Whhd, bihd, bhhd,
                                     Wpk0, Wpk1, WpkD, bias0, bias1, biasD);

    hipMemsetAsync(bar, 0, 256, stream);
    k_lstm<<<16, 256, 0, stream>>>(x, nullptr, 64, 320, Wpk0, (long)10 * 64 * 512, bias0,
                                   h1s, nullptr, nullptr, A0, Gbuf, hstL, cstL, bar, 16, 8);

    hipMemsetAsync(bar, 0, 256, stream);
    k_lstm<<<16, 256, 0, stream>>>(nullptr, h1s, 512, 768, Wpk1, (long)24 * 64 * 512, bias1,
                                   encb, h0c0, h0c0 + 64 * 512, A1, Gbuf, hstL, cstL, bar, 16, 8);

    hipMemsetAsync(bar, 0, 256, stream);
    k_dec<<<64, 512, 0, stream>>>(x, encb, WpkD, biasD, h0c0, AD, Gbuf, hstD, cstD,
                                  h1s /* dec_hs reuses layer0 buffer */, bar, 64);

    k_proj<<<512, 256, 0, stream>>>(h1s, Wp, bp, (float*)d_out);
}
